// HofelLoss_9990093930664
// MI455X (gfx1250) — matrix operand order repaired and checked by reading, not compiled, not run on hardware
//
#include <hip/hip_runtime.h>
#include <math.h>

// ---------------------------------------------------------------------------
// HofelLoss on MI455X (gfx1250): fused split-bf16 WMMA GEMM + hinge epilogue.
//   B = 16384 rows, F = 1024 features.
//   dp - dn = -2*sum_n Y[m,n]*(imgp-imgn)[m,n] + sum_n (imgp^2-imgn^2)[m,n]*colsum[n]
//   (the skt^2 . rowsum term cancels, so it is never computed)
//   Y = skt @ w computed as Ah*Bh + Ah*Bl + Al*Bh in bf16 WMMA, f32 accum.
//   Each wave computes BOTH 16-row subtiles so each B fragment feeds 6 WMMAs,
//   and 4 n-tiles per A fragment so each A fragment feeds 12 WMMAs:
//   per k-tile = 8 ds_b128 + 16 global_b128 + 24 WMMAs -> XDL-limited.
// ---------------------------------------------------------------------------

#define FDIM   1024
#define BROWS  16384
#define DELTA_C 0.3f
#define LAMB_C  0.0005f

typedef __attribute__((ext_vector_type(16))) __bf16 v16bf;
typedef __attribute__((ext_vector_type(8)))  __bf16 v8bf;
typedef __attribute__((ext_vector_type(8)))  float  v8f;
typedef __attribute__((ext_vector_type(4)))  float  v4f;
typedef __attribute__((ext_vector_type(2)))  unsigned int v2u;

union AFrag { v16bf v; v8bf h[2]; };

__device__ __forceinline__ unsigned short f32_to_bf16(float f) {
  unsigned int u = __float_as_uint(f);
  u += 0x7FFFu + ((u >> 16) & 1u);          // round-to-nearest-even
  return (unsigned short)(u >> 16);
}
__device__ __forceinline__ float bf16_bits_to_f32(unsigned short h) {
  return __uint_as_float(((unsigned int)h) << 16);
}

// ---------------------------------------------------------------------------
// Kernel 1: pack w into WMMA-B fragment layout (hi/lo bf16) + regularizer
// partials. Layout per 32x16 (KxN) tile: lane = nin + 16*((kin>>3)&1),
// element = (kin & 7) + 8*(kin>=16)  -> each lane's 16 bf16 are contiguous (32 B).
// Grid: 64 blocks x 256 threads x 64 elems = F*F.
// ---------------------------------------------------------------------------
__global__ void hofel_pack_w(const float* __restrict__ w,
                             const float* __restrict__ eye,
                             unsigned short* __restrict__ whi,
                             unsigned short* __restrict__ wlo,
                             float* __restrict__ l1part,
                             float* __restrict__ l2part) {
  int t = threadIdx.x;
  int base = blockIdx.x * 16384;
  float l1 = 0.f, l2 = 0.f;
  for (int i = 0; i < 64; ++i) {
    int idx = base + t + (i << 8);                 // coalesced
    float val = w[idx];
    float dw  = val - eye[idx];
    l1 += fabsf(dw);
    l2 += dw * dw;
    unsigned short hi = f32_to_bf16(val);
    unsigned short lo = f32_to_bf16(val - bf16_bits_to_f32(hi));
    int k = idx >> 10, n = idx & 1023;             // w[k][n] is B[k][n]
    int kt = k >> 5, nt = n >> 4;
    int kin = k & 31, nin = n & 15;
    int lane = nin + (((kin >> 3) & 1) << 4);
    int off  = (((kt << 6) + nt) << 9) + (lane << 4) + (kin & 7) + ((kin >> 4) << 3);
    whi[off] = hi;
    wlo[off] = lo;
  }
  __shared__ float s1[8], s2[8];
  for (int m = 16; m >= 1; m >>= 1) {
    l1 += __shfl_xor(l1, m, 32);
    l2 += __shfl_xor(l2, m, 32);
  }
  if ((t & 31) == 0) { s1[t >> 5] = l1; s2[t >> 5] = l2; }
  __syncthreads();
  if (t == 0) {
    float a = 0.f, b = 0.f;
    for (int i = 0; i < 8; ++i) { a += s1[i]; b += s2[i]; }
    l1part[blockIdx.x] = a;
    l2part[blockIdx.x] = b;
  }
}

// ---------------------------------------------------------------------------
// Kernel 2: w column sums (column-per-thread => coalesced row-major reads).
// Grid: 4 x 256.
// ---------------------------------------------------------------------------
__global__ void hofel_colsum(const float* __restrict__ w,
                             float* __restrict__ colsum) {
  int j = blockIdx.x * 256 + threadIdx.x;
  float s = 0.f;
  for (int i = 0; i < FDIM; ++i) s += w[i * FDIM + j];
  colsum[j] = s;
}

// ---------------------------------------------------------------------------
// Kernel 3: fused GEMM + hinge epilogue.
// 512 blocks x 256 threads (8 waves). M-tile = 32 rows; every wave computes
// both 16-row subtiles (shared B fragments), 8 n-tiles per wave (2 blocks of 4).
// Dynamic LDS: Ahi[32*1024] + Alo[32*1024] bf16 (128 KB) + Dslot[8][32] f32.
// CDNA5 allows 320 KB/WGP; one workgroup per WGP, 2 waves/SIMD.
// ---------------------------------------------------------------------------
__launch_bounds__(256, 1)
__global__ void hofel_main(const float* __restrict__ skt,
                           const float* __restrict__ imgp,
                           const float* __restrict__ imgn,
                           const unsigned short* __restrict__ whi,
                           const unsigned short* __restrict__ wlo,
                           const float* __restrict__ colsum,
                           float* __restrict__ hpart) {
  extern __shared__ char smem_raw[];
  unsigned short* Ahi = (unsigned short*)smem_raw;      // 32*1024 bf16
  unsigned short* Alo = Ahi + 32 * 1024;                // 32*1024 bf16
  float* Dslot = (float*)(Alo + 32 * 1024);             // [8 waves][32 rows]

  int t = threadIdx.x;
  int rowBlock = blockIdx.x << 5;                       // 32 rows per block

  // ---- Phase 0: stage skt rows as split-bf16 into LDS (b128 in, b64 out) ---
  const float* sbase = skt + (size_t)rowBlock * FDIM;
  for (int i = 0; i < 32; ++i) {
    int e4 = (t + (i << 8)) << 2;                       // element index (x4)
    v4f v = *(const v4f*)(sbase + e4);
    unsigned int h0, h1, l0, l1v;
    {
      unsigned short a0 = f32_to_bf16(v.x), a1 = f32_to_bf16(v.y);
      unsigned short a2 = f32_to_bf16(v.z), a3 = f32_to_bf16(v.w);
      h0 = (unsigned int)a0 | ((unsigned int)a1 << 16);
      h1 = (unsigned int)a2 | ((unsigned int)a3 << 16);
      unsigned short b0 = f32_to_bf16(v.x - bf16_bits_to_f32(a0));
      unsigned short b1 = f32_to_bf16(v.y - bf16_bits_to_f32(a1));
      unsigned short b2 = f32_to_bf16(v.z - bf16_bits_to_f32(a2));
      unsigned short b3 = f32_to_bf16(v.w - bf16_bits_to_f32(a3));
      l0 = (unsigned int)b0 | ((unsigned int)b1 << 16);
      l1v = (unsigned int)b2 | ((unsigned int)b3 << 16);
    }
    *(v2u*)(Ahi + e4) = (v2u){h0, h1};
    *(v2u*)(Alo + e4) = (v2u){l0, l1v};
  }
  __syncthreads();

  int wave = t >> 5, lane = t & 31;
  int l15 = lane & 15;
  int khalf = (lane >> 4) << 3;    // A-fragment K base: lanes>=16 start at K+8
  int mhalf = khalf;               // C/D rows: lanes>=16 hold M = r+8

  // Subtile 0 = rows 0..15, subtile 1 = rows 16..31 (same for every wave).
  const unsigned short* a0_hi = Ahi + l15 * FDIM;
  const unsigned short* a0_lo = Alo + l15 * FDIM;
  const unsigned short* a1_hi = Ahi + (16 + l15) * FDIM;
  const unsigned short* a1_lo = Alo + (16 + l15) * FDIM;

  float d0[8], d1[8];
#pragma unroll
  for (int r = 0; r < 8; ++r) { d0[r] = 0.f; d1[r] = 0.f; }

  for (int jblk = 0; jblk < 2; ++jblk) {
    int ntbase = (wave << 3) + (jblk << 2);             // 4 n-tiles per block
    const unsigned short* bpb = whi + ((size_t)ntbase << 9) + (lane << 4);
    const unsigned short* bqb = wlo + ((size_t)ntbase << 9) + (lane << 4);
    v8f c0[4], c1[4];
#pragma unroll
    for (int j2 = 0; j2 < 4; ++j2) { c0[j2] = (v8f){}; c1[j2] = (v8f){}; }

#pragma unroll 2
    for (int kt = 0; kt < 32; ++kt) {
      int k0 = kt << 5;
      // Two A-fragment pairs (both subtiles) -> 24 WMMAs per k-tile.
      AFrag ah0, al0, ah1, al1;
      ah0.h[0] = *(const v8bf*)(a0_hi + k0 + khalf);
      ah0.h[1] = *(const v8bf*)(a0_hi + k0 + khalf + 16);
      al0.h[0] = *(const v8bf*)(a0_lo + k0 + khalf);
      al0.h[1] = *(const v8bf*)(a0_lo + k0 + khalf + 16);
      ah1.h[0] = *(const v8bf*)(a1_hi + k0 + khalf);
      ah1.h[1] = *(const v8bf*)(a1_hi + k0 + khalf + 16);
      al1.h[0] = *(const v8bf*)(a1_lo + k0 + khalf);
      al1.h[1] = *(const v8bf*)(a1_lo + k0 + khalf + 16);
      __builtin_prefetch(bpb + (((size_t)kt + 1) << 15), 0, 3);
#pragma unroll
      for (int j2 = 0; j2 < 4; ++j2) {
        size_t boff = ((size_t)kt << 15) + ((size_t)j2 << 9);
        v16bf bh = *(const v16bf*)(bpb + boff);
        v16bf bl = *(const v16bf*)(bqb + boff);
        // Each B pair feeds both subtiles: 6 WMMAs per 4 global_b128.
        c0[j2] = __builtin_amdgcn_wmma_f32_16x16x32_bf16(
                     false, ah0.v, false, bh, (short)0, c0[j2], false, false);
        c1[j2] = __builtin_amdgcn_wmma_f32_16x16x32_bf16(
                     false, ah1.v, false, bh, (short)0, c1[j2], false, false);
        c0[j2] = __builtin_amdgcn_wmma_f32_16x16x32_bf16(
                     false, ah0.v, false, bl, (short)0, c0[j2], false, false);
        c1[j2] = __builtin_amdgcn_wmma_f32_16x16x32_bf16(
                     false, ah1.v, false, bl, (short)0, c1[j2], false, false);
        c0[j2] = __builtin_amdgcn_wmma_f32_16x16x32_bf16(
                     false, al0.v, false, bh, (short)0, c0[j2], false, false);
        c1[j2] = __builtin_amdgcn_wmma_f32_16x16x32_bf16(
                     false, al1.v, false, bh, (short)0, c1[j2], false, false);
      }
    }
    // ---- Epilogue: fold imgp/imgn straight into per-row accumulators ----
    // C layout: VGPR r, lanes 0-15 -> (M=r, N=lane); lanes 16-31 -> (M=r+8).
#pragma unroll
    for (int j2 = 0; j2 < 4; ++j2) {
      int nb = (ntbase + j2) << 4;
      float wcs = colsum[nb + l15];
      size_t g0 = (size_t)(rowBlock + mhalf) * FDIM + nb + l15;
      size_t g1 = (size_t)(rowBlock + 16 + mhalf) * FDIM + nb + l15;
#pragma unroll
      for (int r = 0; r < 8; ++r) {
        float p0  = imgp[g0 + (size_t)r * FDIM];
        float n0  = imgn[g0 + (size_t)r * FDIM];
        d0[r] += (p0 - n0) * ((p0 + n0) * wcs - 2.0f * c0[j2][r]);
        float p1  = imgp[g1 + (size_t)r * FDIM];
        float n1  = imgn[g1 + (size_t)r * FDIM];
        d1[r] += (p1 - n1) * ((p1 + n1) * wcs - 2.0f * c1[j2][r]);
      }
    }
  }

  // Reduce each d[r] across its 16-lane half (xor masks stay within halves).
#pragma unroll
  for (int r = 0; r < 8; ++r) {
    for (int m = 8; m >= 1; m >>= 1) {
      d0[r] += __shfl_xor(d0[r], m, 32);
      d1[r] += __shfl_xor(d1[r], m, 32);
    }
  }
  if (l15 == 0) {
    int rbase = (wave << 5) + mhalf;   // fixed per-wave slot: deterministic
#pragma unroll
    for (int r = 0; r < 8; ++r) {
      Dslot[rbase + r]      = d0[r];
      Dslot[rbase + 16 + r] = d1[r];
    }
  }
  __syncthreads();

  // ---- Phase 2: hinge per row, deterministic fixed-order combine ----
  if (t < 32) {
    float D = 0.f;
#pragma unroll
    for (int wv = 0; wv < 8; ++wv) D += Dslot[(wv << 5) + t];
    float h = fmaxf(DELTA_C + D, 0.f);
    for (int m = 16; m >= 1; m >>= 1) h += __shfl_xor(h, m, 32);
    if (t == 0) hpart[blockIdx.x] = h;
  }
}

// ---------------------------------------------------------------------------
// Kernel 4: final deterministic reduction -> scalar loss.
// ---------------------------------------------------------------------------
__global__ void hofel_final(const float* __restrict__ hpart,
                            const float* __restrict__ l1part,
                            const float* __restrict__ l2part,
                            float* __restrict__ out) {
  int t = threadIdx.x;
  float h  = hpart[t] + hpart[t + 256];
  float l1 = (t < 64) ? l1part[t] : 0.f;
  float l2 = (t < 64) ? l2part[t] : 0.f;
  __shared__ float sh[8], s1[8], s2[8];
  for (int m = 16; m >= 1; m >>= 1) {
    h  += __shfl_xor(h,  m, 32);
    l1 += __shfl_xor(l1, m, 32);
    l2 += __shfl_xor(l2, m, 32);
  }
  if ((t & 31) == 0) { sh[t >> 5] = h; s1[t >> 5] = l1; s2[t >> 5] = l2; }
  __syncthreads();
  if (t == 0) {
    float H = 0.f, A = 0.f, S = 0.f;
    for (int i = 0; i < 8; ++i) { H += sh[i]; A += s1[i]; S += s2[i]; }
    out[0] = H / (float)BROWS + LAMB_C * (A + sqrtf(S));
  }
}

// ---------------------------------------------------------------------------
extern "C" void kernel_launch(void* const* d_in, const int* in_sizes, int n_in,
                              void* d_out, int out_size, void* d_ws,
                              size_t ws_size, hipStream_t stream) {
  (void)in_sizes; (void)n_in; (void)out_size; (void)ws_size;
  const float* skt  = (const float*)d_in[0];
  const float* imgp = (const float*)d_in[1];
  const float* imgn = (const float*)d_in[2];
  const float* w    = (const float*)d_in[3];
  const float* eye  = (const float*)d_in[4];
  float* out = (float*)d_out;

  char* ws = (char*)d_ws;                         // ~4.2 MB used
  unsigned short* whi = (unsigned short*)(ws);                 // 2 MB
  unsigned short* wlo = (unsigned short*)(ws + (2u << 20));    // 2 MB
  float* colsum = (float*)(ws + (4u << 20));                   // 4 KB
  float* hpart  = (float*)(ws + (4u << 20) + 4096);            // 2 KB
  float* l1part = (float*)(ws + (4u << 20) + 4096 + 2048);     // 256 B
  float* l2part = (float*)(ws + (4u << 20) + 4096 + 2048 + 256);

  hofel_pack_w<<<64, 256, 0, stream>>>(w, eye, whi, wlo, l1part, l2part);
  hofel_colsum<<<4, 256, 0, stream>>>(w, colsum);

  size_t smem = 2u * 32 * 1024 * sizeof(unsigned short) +   // Ahi + Alo
                8u * 32 * sizeof(float);                     // Dslot[8][32]
  hofel_main<<<BROWS / 32, 256, smem, stream>>>(skt, imgp, imgn, whi, wlo,
                                                colsum, hpart);
  hofel_final<<<1, 256, 0, stream>>>(hpart, l1part, l2part, out);
}
